// GCN_34969623724071
// MI455X (gfx1250) — matrix operand order repaired and checked by reading, not compiled, not run on hardware
//
#include <hip/hip_runtime.h>
#include <hip/hip_bf16.h>

typedef __attribute__((ext_vector_type(16))) _Float16 v16h;
typedef __attribute__((ext_vector_type(8)))  _Float16 v8h;
typedef __attribute__((ext_vector_type(8)))  float    v8f;

#define FEAT 128

// ---------------- elementwise helpers ----------------
__global__ void zero_f32(float* __restrict__ p, int n) {
  int i = blockIdx.x * 256 + threadIdx.x;
  if (i < n) p[i] = 0.f;
}

__global__ void cvt_f32_f16(const float* __restrict__ in, _Float16* __restrict__ out, int n) {
  int i = blockIdx.x * 256 + threadIdx.x;
  if (i < n) out[i] = (_Float16)in[i];
}

// W [K=128, N=128] row-major f32 -> Wt [N, K] f16 (column-major of W)
__global__ void w_transpose_f16(const float* __restrict__ W, _Float16* __restrict__ Wt) {
  int k = blockIdx.x, n = threadIdx.x;
  Wt[n * FEAT + k] = (_Float16)W[k * FEAT + n];
}

__global__ void bias_relu(float* __restrict__ a, const float* __restrict__ b, int n) {
  int i = blockIdx.x * 256 + threadIdx.x;
  if (i < n) {
    float v = a[i] + b[i & (FEAT - 1)];
    a[i] = v > 0.f ? v : 0.f;
  }
}

// ---------------- WMMA GEMM: C[M,128] = A[M,128](f16) @ Wt'[128,128] ----------------
// Wt is stored [n][k] (f16) so B-fragment loads are contiguous.
// One wave per 16x16 tile; 8 waves/block cover all 128 output columns.
__global__ __launch_bounds__(256) void gemm_wmma_f16(
    const _Float16* __restrict__ A, const _Float16* __restrict__ Wt,
    float* __restrict__ C, int Nrows) {
  const int wave = threadIdx.x >> 5;
  const int lane = threadIdx.x & 31;
  const int m    = lane & 15;
  const int hi   = lane >> 4;
  const int row0 = blockIdx.x * 16;
  const int col0 = wave * 16;
  (void)Nrows;

  const _Float16* arow = A  + (size_t)(row0 + m) * FEAT;
  const _Float16* bcol = Wt + (size_t)(col0 + m) * FEAT;

  v8f acc = {};
#pragma unroll
  for (int kb = 0; kb < FEAT; kb += 32) {
    // A fragment, ISA 16-bit A 16x32 layout:
    //  halves 0..7  : K = kb + hi*8 + h
    //  halves 8..15 : K = kb + 16 + hi*8 + (h-8)
    v8h alo = *(const v8h*)(arow + kb + hi * 8);
    v8h ahi = *(const v8h*)(arow + kb + 16 + hi * 8);
    v16h av = __builtin_shufflevector(alo, ahi, 0,1,2,3,4,5,6,7,8,9,10,11,12,13,14,15);
    // B fragment: column n = col0+m, same K layout as A (contiguous in Wt)
    v8h blo = *(const v8h*)(bcol + kb + hi * 8);
    v8h bhi = *(const v8h*)(bcol + kb + 16 + hi * 8);
    v16h bv = __builtin_shufflevector(blo, bhi, 0,1,2,3,4,5,6,7,8,9,10,11,12,13,14,15);
    acc = __builtin_amdgcn_wmma_f32_16x16x32_f16(false, av, false, bv,
                                                 (short)0, acc, false, false);
  }
  // D layout: lanes 0-15 VGPR r -> M=r ; lanes 16-31 -> M=r+8
#pragma unroll
  for (int r = 0; r < 8; ++r)
    C[(size_t)(row0 + r + 8 * hi) * FEAT + col0 + m] = acc[r];
}

// ---------------- edge aggregation: agg[dst] += sup[src] (masked) ----------------
// 32 threads per edge, 4 floats each (float4 gather + 4 f32 atomics).
__global__ void aggregate_edges(const float* __restrict__ sup,
                                const int* __restrict__ src,
                                const int* __restrict__ dst,
                                const unsigned char* __restrict__ em,
                                float* __restrict__ agg, int E) {
  int t = blockIdx.x * 256 + threadIdx.x;
  int e = t >> 5;
  if (e >= E) return;
  if (em && !em[e]) return;
  int q = (t & 31) * 4;
  const float4 v = *(const float4*)(sup + (size_t)src[e] * FEAT + q);
  float* d = agg + (size_t)dst[e] * FEAT + q;
  atomicAdd(d + 0, v.x);
  atomicAdd(d + 1, v.y);
  atomicAdd(d + 2, v.z);
  atomicAdd(d + 3, v.w);
}

// ---------------- pooling: ||p||, scores, top-k, edge remap ----------------
__global__ void pnorm_kernel(const float* __restrict__ p, float* __restrict__ pn) {
  __shared__ float s[FEAT];
  int t = threadIdx.x;
  float v = p[t];
  s[t] = v * v;
  __syncthreads();
  for (int o = 64; o; o >>= 1) {
    if (t < o) s[t] += s[t + o];
    __syncthreads();
  }
  if (t == 0) pn[0] = sqrtf(s[0]);
}

// one wave per node: score = tanh(dot(h,p)/||p||)
__global__ void score_kernel(const float* __restrict__ h, const float* __restrict__ p,
                             const float* __restrict__ pn, float* __restrict__ score,
                             int Nn) {
  int t = blockIdx.x * 256 + threadIdx.x;
  int node = t >> 5, lane = t & 31;
  if (node >= Nn) return;
  float s = 0.f;
  const float* hr = h + (size_t)node * FEAT;
#pragma unroll
  for (int j = 0; j < FEAT; j += 32) s += hr[j + lane] * p[j + lane];
  for (int off = 16; off; off >>= 1) s += __shfl_xor(s, off, 32);
  if (lane == 0) score[node] = tanhf(s / pn[0]);
}

// one block (n threads) per graph; rank-select top-k, gate by score, build node_map
__global__ void topk_kernel(const float* __restrict__ h, const float* __restrict__ score,
                            float* __restrict__ hout, int* __restrict__ node_map,
                            int n, int k) {
  __shared__ float s[512];
  int g = blockIdx.x, i = threadIdx.x;
  int gid = g * n + i;
  float my = score[gid];
  s[i] = my;
  __syncthreads();
  int rank = 0;
  for (int j = 0; j < n; ++j) {
    float sj = s[j];
    rank += (sj > my) || (sj == my && j < i);
  }
  if (rank < k) {
    int ng = g * k + rank;
    node_map[gid] = ng;
    const float* hr = h + (size_t)gid * FEAT;
    float* ho = hout + (size_t)ng * FEAT;
#pragma unroll 4
    for (int f = 0; f < FEAT; f += 4) {
      float4 v = *(const float4*)(hr + f);
      v.x *= my; v.y *= my; v.z *= my; v.w *= my;
      *(float4*)(ho + f) = v;
    }
  } else {
    node_map[gid] = -1;
  }
}

__global__ void remap_edges(const int* __restrict__ s, const int* __restrict__ d,
                            const unsigned char* __restrict__ emOld,
                            const int* __restrict__ node_map,
                            int* __restrict__ sNew, int* __restrict__ dNew,
                            unsigned char* __restrict__ emNew, int E) {
  int e = blockIdx.x * 256 + threadIdx.x;
  if (e >= E) return;
  int ms = node_map[s[e]];
  int md = node_map[d[e]];
  unsigned char keep = emOld ? emOld[e] : (unsigned char)1;
  keep = (unsigned char)(keep && (ms >= 0) && (md >= 0));
  sNew[e] = ms >= 0 ? ms : 0;
  dNew[e] = md >= 0 ? md : 0;
  emNew[e] = keep;
}

// ---------------- readout: out[g] (+)= [max | mean] over k nodes ----------------
__global__ void readout_kernel(const float* __restrict__ h, float* __restrict__ out,
                               int k, int accumulate) {
  int g = blockIdx.x, f = threadIdx.x;  // 128 threads
  const float* base = h + (size_t)g * k * FEAT + f;
  float mx = -__builtin_inff(), sm = 0.f;
  for (int i = 0; i < k; ++i) {
    float v = base[(size_t)i * FEAT];
    mx = fmaxf(mx, v);
    sm += v;
  }
  float mean = sm / (float)k;
  float* o = out + (size_t)g * 256;
  if (accumulate) { o[f] += mx; o[FEAT + f] += mean; }
  else            { o[f]  = mx; o[FEAT + f]  = mean; }
}

// ---------------- driver ----------------
extern "C" void kernel_launch(void* const* d_in, const int* in_sizes, int n_in,
                              void* d_out, int out_size, void* d_ws, size_t ws_size,
                              hipStream_t stream) {
  const int BGRAPH = 256, NPER = 512;
  const int N0 = BGRAPH * NPER;       // 131072
  const int K1 = 256, N1 = BGRAPH * K1;   // 65536
  const int K2 = 128, N2 = BGRAPH * K2;   // 32768
  const int E = in_sizes[1];

  const float* x  = (const float*)d_in[0];
  const int* src  = (const int*)d_in[1];
  const int* dst  = (const int*)d_in[2];
  const float* W1 = (const float*)d_in[4];
  const float* b1 = (const float*)d_in[5];
  const float* W2 = (const float*)d_in[6];
  const float* b2 = (const float*)d_in[7];
  const float* W3 = (const float*)d_in[8];
  const float* b3 = (const float*)d_in[9];
  const float* p1 = (const float*)d_in[10];
  const float* p2 = (const float*)d_in[11];
  float* out = (float*)d_out;

  // workspace layout
  char* ws = (char*)d_ws;
  size_t off = 0;
  auto alloc = [&](size_t bytes) -> void* {
    void* p = ws + off;
    off = (off + bytes + 255) & ~(size_t)255;
    return p;
  };
  float*        S      = (float*)alloc((size_t)N0 * FEAT * 4);   // GEMM output (support)
  float*        Abuf   = (float*)alloc((size_t)N0 * FEAT * 4);   // agg -> h (in place), reused per level
  float*        Bbuf   = (float*)alloc((size_t)N1 * FEAT * 4);   // pooled h, reused per level
  _Float16*     Xh     = (_Float16*)alloc((size_t)N0 * FEAT * 2);
  _Float16*     Wt1    = (_Float16*)alloc((size_t)FEAT * FEAT * 2);
  _Float16*     Wt2    = (_Float16*)alloc((size_t)FEAT * FEAT * 2);
  _Float16*     Wt3    = (_Float16*)alloc((size_t)FEAT * FEAT * 2);
  float*        scores = (float*)alloc((size_t)N0 * 4);
  int*          nmap   = (int*)alloc((size_t)N0 * 4);
  int*          src1   = (int*)alloc((size_t)E * 4);
  int*          dst1   = (int*)alloc((size_t)E * 4);
  int*          src2   = (int*)alloc((size_t)E * 4);
  int*          dst2   = (int*)alloc((size_t)E * 4);
  unsigned char* em1   = (unsigned char*)alloc((size_t)E);
  unsigned char* em2   = (unsigned char*)alloc((size_t)E);
  float*        pn     = (float*)alloc(256);
  (void)ws_size; (void)n_in; (void)out_size;

  const int elemsN0 = N0 * FEAT, elemsN1 = N1 * FEAT, elemsN2 = N2 * FEAT;
  const int eThreads = E * 32;

  // weight prep
  w_transpose_f16<<<FEAT, FEAT, 0, stream>>>(W1, Wt1);
  w_transpose_f16<<<FEAT, FEAT, 0, stream>>>(W2, Wt2);
  w_transpose_f16<<<FEAT, FEAT, 0, stream>>>(W3, Wt3);

  // ===== level 0: gconv1 =====
  cvt_f32_f16<<<elemsN0 / 256, 256, 0, stream>>>(x, Xh, elemsN0);
  gemm_wmma_f16<<<N0 / 16, 256, 0, stream>>>(Xh, Wt1, S, N0);
  zero_f32<<<elemsN0 / 256, 256, 0, stream>>>(Abuf, elemsN0);
  aggregate_edges<<<(eThreads + 255) / 256, 256, 0, stream>>>(S, src, dst, nullptr, Abuf, E);
  bias_relu<<<elemsN0 / 256, 256, 0, stream>>>(Abuf, b1, elemsN0);   // h0 = Abuf

  // pool 1
  pnorm_kernel<<<1, FEAT, 0, stream>>>(p1, pn);
  score_kernel<<<(N0 * 32) / 256, 256, 0, stream>>>(Abuf, p1, pn, scores, N0);
  topk_kernel<<<BGRAPH, NPER, 0, stream>>>(Abuf, scores, Bbuf, nmap, NPER, K1);  // h1 = Bbuf
  remap_edges<<<(E + 255) / 256, 256, 0, stream>>>(src, dst, nullptr, nmap, src1, dst1, em1, E);
  readout_kernel<<<BGRAPH, FEAT, 0, stream>>>(Bbuf, out, K1, 0);     // x1 (=)

  // ===== level 1: gconv2 =====
  cvt_f32_f16<<<elemsN1 / 256, 256, 0, stream>>>(Bbuf, Xh, elemsN1);
  gemm_wmma_f16<<<N1 / 16, 256, 0, stream>>>(Xh, Wt2, S, N1);
  zero_f32<<<elemsN1 / 256, 256, 0, stream>>>(Abuf, elemsN1);
  aggregate_edges<<<(eThreads + 255) / 256, 256, 0, stream>>>(S, src1, dst1, em1, Abuf, E);
  bias_relu<<<elemsN1 / 256, 256, 0, stream>>>(Abuf, b2, elemsN1);   // h1b = Abuf

  // pool 2
  pnorm_kernel<<<1, FEAT, 0, stream>>>(p2, pn);
  score_kernel<<<(N1 * 32) / 256, 256, 0, stream>>>(Abuf, p2, pn, scores, N1);
  topk_kernel<<<BGRAPH, K1, 0, stream>>>(Abuf, scores, Bbuf, nmap, K1, K2);      // h2 = Bbuf
  remap_edges<<<(E + 255) / 256, 256, 0, stream>>>(src1, dst1, em1, nmap, src2, dst2, em2, E);
  readout_kernel<<<BGRAPH, FEAT, 0, stream>>>(Bbuf, out, K2, 1);     // x2 (+=)

  // ===== level 2: gconv3 =====
  cvt_f32_f16<<<elemsN2 / 256, 256, 0, stream>>>(Bbuf, Xh, elemsN2);
  gemm_wmma_f16<<<N2 / 16, 256, 0, stream>>>(Xh, Wt3, S, N2);
  zero_f32<<<elemsN2 / 256, 256, 0, stream>>>(Abuf, elemsN2);
  aggregate_edges<<<(eThreads + 255) / 256, 256, 0, stream>>>(S, src2, dst2, em2, Abuf, E);
  bias_relu<<<elemsN2 / 256, 256, 0, stream>>>(Abuf, b3, elemsN2);   // h2b = Abuf
  readout_kernel<<<BGRAPH, FEAT, 0, stream>>>(Abuf, out, K2, 1);     // x3 (+=)
}
